// BiAttention_84456236908627
// MI455X (gfx1250) — matrix operand order repaired and checked by reading, not compiled, not run on hardware
//
#include <hip/hip_runtime.h>

typedef __attribute__((ext_vector_type(16))) __bf16 v16bf;
typedef __attribute__((ext_vector_type(8)))  float  v8f;

union Frag {
  v16bf v;
  __bf16 e[16];
};

#define DEVINL __device__ __forceinline__

static constexpr int   SEQ   = 2048;        // Lp == Lh
static constexpr int   DIM   = 256;         // D
static constexpr int   NCH   = DIM / 32;    // 8 K-chunks of 32 for bf16 WMMA
static constexpr int   KB    = 32;          // key-side rows staged in LDS per step
static constexpr int   WAVES = 4;           // 128-thread workgroup (wave32)
static constexpr int   ROWB  = 16 * WAVES;  // 64 query rows per workgroup
static constexpr int   SH    = 272;         // LDS row stride (272*2B = 544 = 17*32 -> v16bf stays 32B-aligned)
static constexpr float SCALE = 0.0625f;     // 1/sqrt(256)
static constexpr float NEGI  = -1e30f;
static constexpr float EPS   = 1e-13f;

// Native bf16 hi/lo split: hi = rne(x), lo = rne(x - hi).  Lowered to v_cvt_*bf16* ops.
DEVINL void split_bf16(float x, __bf16& hi, __bf16& lo) {
  hi = (__bf16)x;
  lo = (__bf16)(x - (float)hi);
}

DEVINL v8f wmma_bf16(const Frag& a, const Frag& b, v8f c) {
  // D = A(16x32 bf16) x B(32x16 bf16) + C(16x16 f32)
  return __builtin_amdgcn_wmma_f32_16x16x32_bf16(false, a.v, false, b.v, (short)0, c, false, false);
}

// Build the wave's 16x256 A-matrix fragments (hi/lo bf16 split) straight from global.
// A order used here (B matches): lane holds row (lane&15); per 32-chunk the lane's K list is
// {0..15} + 16*(lane>>4)  -> two contiguous groups of 8 floats.
DEVINL void load_A(const float* __restrict__ row_ptr, int lane, Frag* ahi, Frag* alo) {
  const int h8 = 16 * (lane >> 4);
#pragma unroll
  for (int ch = 0; ch < NCH; ++ch) {
    const int k0 = ch * 32 + h8;
#pragma unroll
    for (int g = 0; g < 2; ++g) {
      const float4* p = (const float4*)(row_ptr + k0 + 8 * g);
      float4 x = p[0];
      float4 y = p[1];
      float vals[8] = {x.x, x.y, x.z, x.w, y.x, y.y, y.z, y.w};
#pragma unroll
      for (int i = 0; i < 8; ++i)
        split_bf16(vals[i], ahi[ch].e[g * 8 + i], alo[ch].e[g * 8 + i]);
    }
  }
}

// Cooperatively stage a KB x DIM fp32 tile of the key-side matrix into LDS as bf16 hi/lo
// (row-major, for score B-fragments) and optionally transposed hi (for value B-fragments).
DEVINL void stage_K(const float* __restrict__ Kb, int k0, int tid,
                    __bf16 (*Hhi)[SH], __bf16 (*Hlo)[SH],
                    __bf16 (*VT)[KB]) {
#pragma unroll
  for (int it = 0; it < (KB * DIM) / (WAVES * 32 * 4); ++it) {  // 16 float4 per thread
    int idx = (it * (WAVES * 32) + tid) * 4;
    int r = idx >> 8;           // / DIM
    int c = idx & (DIM - 1);
    float4 x = *(const float4*)(Kb + (size_t)(k0 + r) * DIM + c);
    float vals[4] = {x.x, x.y, x.z, x.w};
#pragma unroll
    for (int i = 0; i < 4; ++i) {
      __bf16 hb, lb;
      split_bf16(vals[i], hb, lb);
      Hhi[r][c + i] = hb;
      Hlo[r][c + i] = lb;
      if (VT) VT[c + i][r] = hb;
    }
  }
}

// ---------------- Pass 1: per-row softmax statistics (max, sum of exp) ----------------
__global__ __launch_bounds__(128) void esim_stats_kernel(
    const float* __restrict__ Q, const float* __restrict__ Kmat,
    const int* __restrict__ kmask,
    float* __restrict__ rowmax, float* __restrict__ rowsum) {
  __shared__ __align__(32) __bf16 Hhi[KB][SH];
  __shared__ __align__(32) __bf16 Hlo[KB][SH];

  const int b    = blockIdx.y;
  const int tid  = threadIdx.x;
  const int wave = tid >> 5;
  const int lane = tid & 31;
  const int h    = lane >> 4;
  const int nlo  = lane & 15;
  const int qrow0 = blockIdx.x * ROWB + wave * 16;

  const float* Qb = Q    + (size_t)b * SEQ * DIM;
  const float* Kb = Kmat + (size_t)b * SEQ * DIM;
  const int*   km = kmask + (size_t)b * SEQ;

  Frag ahi[NCH], alo[NCH];
  load_A(Qb + (size_t)(qrow0 + nlo) * DIM, lane, ahi, alo);

  float m[8], s[8];
#pragma unroll
  for (int r = 0; r < 8; ++r) { m[r] = NEGI; s[r] = 0.f; }

  for (int k0 = 0; k0 < SEQ; k0 += KB) {
    if (k0 + KB < SEQ)
      __builtin_prefetch(Kb + (size_t)(k0 + KB) * DIM + tid * 16, 0, 0);
    stage_K(Kb, k0, tid, Hhi, Hlo, nullptr);
    __syncthreads();
#pragma unroll
    for (int nt = 0; nt < KB / 16; ++nt) {
      const int colloc = nt * 16 + nlo;
      v8f acc;
#pragma unroll
      for (int i = 0; i < 8; ++i) acc[i] = 0.f;
#pragma unroll
      for (int ch = 0; ch < NCH; ++ch) {
        Frag bh, bl;
        // B order used here (matches A): lane holds column (lane&15); K list {0..15}+16*(lane>>4) -> one 32B read
        bh.v = *(const v16bf*)&Hhi[colloc][ch * 32 + 16 * h];
        bl.v = *(const v16bf*)&Hlo[colloc][ch * 32 + 16 * h];
        acc = wmma_bf16(ahi[ch], bh, acc);   // hi*hi
        acc = wmma_bf16(alo[ch], bh, acc);   // lo*hi
        acc = wmma_bf16(ahi[ch], bl, acc);   // hi*lo
      }
      const float maskv = (float)km[k0 + colloc];
#pragma unroll
      for (int r = 0; r < 8; ++r) {
        float sc = (maskv != 0.f) ? acc[r] * SCALE : NEGI;
        float tm = sc;                       // max over the 16 lanes of this half
        tm = fmaxf(tm, __shfl_xor(tm, 1));
        tm = fmaxf(tm, __shfl_xor(tm, 2));
        tm = fmaxf(tm, __shfl_xor(tm, 4));
        tm = fmaxf(tm, __shfl_xor(tm, 8));
        float nm = fmaxf(m[r], tm);
        float e = __expf(sc - nm);
        e += __shfl_xor(e, 1);
        e += __shfl_xor(e, 2);
        e += __shfl_xor(e, 4);
        e += __shfl_xor(e, 8);
        s[r] = s[r] * __expf(m[r] - nm) + e; // online rescale
        m[r] = nm;
      }
    }
    __syncthreads();
  }

  if (nlo == 0) {
#pragma unroll
    for (int r = 0; r < 8; ++r) {
      int row = qrow0 + r + 8 * h;
      rowmax[(size_t)b * SEQ + row] = m[r];
      rowsum[(size_t)b * SEQ + row] = s[r];
    }
  }
}

// ---------------- Pass 2: attended output O = P(softmax) x V, masked on query side ----------------
__global__ __launch_bounds__(128) void esim_attend_kernel(
    const float* __restrict__ Q, const float* __restrict__ Kmat,
    const int* __restrict__ qmask, const int* __restrict__ kmask,
    const float* __restrict__ rowmax, const float* __restrict__ rowsum,
    float* __restrict__ Out) {
  __shared__ __align__(32) __bf16 Hhi[KB][SH];
  __shared__ __align__(32) __bf16 Hlo[KB][SH];
  __shared__ __align__(32) __bf16 VT[DIM][KB];           // transposed value tile (hi bf16)
  __shared__ __align__(16) float Pbuf[WAVES][16][36];    // per-wave P staging (f32)

  const int b    = blockIdx.y;
  const int tid  = threadIdx.x;
  const int wave = tid >> 5;
  const int lane = tid & 31;
  const int h    = lane >> 4;
  const int nlo  = lane & 15;
  const int qrow0 = blockIdx.x * ROWB + wave * 16;

  const float* Qb = Q    + (size_t)b * SEQ * DIM;
  const float* Kb = Kmat + (size_t)b * SEQ * DIM;
  const int*   km = kmask + (size_t)b * SEQ;

  Frag ahi[NCH], alo[NCH];
  load_A(Qb + (size_t)(qrow0 + nlo) * DIM, lane, ahi, alo);

  float M[8], invL[8];
#pragma unroll
  for (int r = 0; r < 8; ++r) {
    int row = qrow0 + r + 8 * h;
    M[r]    = rowmax[(size_t)b * SEQ + row];
    invL[r] = 1.0f / (rowsum[(size_t)b * SEQ + row] + EPS);
  }

  v8f o[DIM / 16];
#pragma unroll
  for (int ot = 0; ot < DIM / 16; ++ot)
#pragma unroll
    for (int i = 0; i < 8; ++i) o[ot][i] = 0.f;

  for (int k0 = 0; k0 < SEQ; k0 += KB) {
    if (k0 + KB < SEQ)
      __builtin_prefetch(Kb + (size_t)(k0 + KB) * DIM + tid * 16, 0, 0);
    stage_K(Kb, k0, tid, Hhi, Hlo, VT);
    __syncthreads();

    // --- scores for this 16x32 block, then P into per-wave LDS ---
#pragma unroll
    for (int nt = 0; nt < KB / 16; ++nt) {
      const int colloc = nt * 16 + nlo;
      v8f acc;
#pragma unroll
      for (int i = 0; i < 8; ++i) acc[i] = 0.f;
#pragma unroll
      for (int ch = 0; ch < NCH; ++ch) {
        Frag bh, bl;
        bh.v = *(const v16bf*)&Hhi[colloc][ch * 32 + 16 * h];
        bl.v = *(const v16bf*)&Hlo[colloc][ch * 32 + 16 * h];
        acc = wmma_bf16(ahi[ch], bh, acc);
        acc = wmma_bf16(alo[ch], bh, acc);
        acc = wmma_bf16(ahi[ch], bl, acc);
      }
      const float maskv = (float)km[k0 + colloc];
#pragma unroll
      for (int r = 0; r < 8; ++r) {
        float sc = (maskv != 0.f) ? acc[r] * SCALE : NEGI;
        float p  = maskv * __expf(sc - M[r]) * invL[r];
        Pbuf[wave][r + 8 * h][colloc] = p;
      }
    }
    // same-wave LDS producer/consumer: DS ops are in-order within a wave, no barrier needed

    // --- P as A-fragment (hi/lo split), K = 32 == this whole key tile ---
    Frag phi, plo;
    {
      const float* pr = &Pbuf[wave][nlo][0];
      const int h8 = 16 * h;
#pragma unroll
      for (int g = 0; g < 2; ++g) {
        const float4* p4 = (const float4*)(pr + h8 + 8 * g);
        float4 x = p4[0];
        float4 y = p4[1];
        float vals[8] = {x.x, x.y, x.z, x.w, y.x, y.y, y.z, y.w};
#pragma unroll
        for (int i = 0; i < 8; ++i)
          split_bf16(vals[i], phi.e[g * 8 + i], plo.e[g * 8 + i]);
      }
    }

    // --- O += P x V over all 16 output column tiles ---
#pragma unroll
    for (int ot = 0; ot < DIM / 16; ++ot) {
      Frag vb;
      vb.v = *(const v16bf*)&VT[ot * 16 + nlo][16 * h];
      o[ot] = wmma_bf16(phi, vb, o[ot]);
      o[ot] = wmma_bf16(plo, vb, o[ot]);
    }
    __syncthreads();
  }

  // epilogue: query-side mask and store
  float qm[8];
#pragma unroll
  for (int r = 0; r < 8; ++r)
    qm[r] = (float)qmask[(size_t)b * SEQ + qrow0 + r + 8 * h];
#pragma unroll
  for (int ot = 0; ot < DIM / 16; ++ot)
#pragma unroll
    for (int r = 0; r < 8; ++r) {
      int row = qrow0 + r + 8 * h;
      Out[((size_t)b * SEQ + row) * DIM + ot * 16 + nlo] = o[ot][r] * qm[r];
    }
}

extern "C" void kernel_launch(void* const* d_in, const int* in_sizes, int n_in,
                              void* d_out, int out_size, void* d_ws, size_t ws_size,
                              hipStream_t stream) {
  (void)in_sizes; (void)n_in; (void)out_size; (void)ws_size;
  const float* P  = (const float*)d_in[0];   // premise_batch   (16,2048,256) f32
  const int*   pm = (const int*)d_in[1];     // premise_mask    (16,2048) i32
  const float* H  = (const float*)d_in[2];   // hypothesis_batch(16,2048,256) f32
  const int*   hm = (const int*)d_in[3];     // hypothesis_mask (16,2048) i32

  constexpr int B = 16;
  float* outP = (float*)d_out;                           // attended_premises
  float* outH = outP + (size_t)B * SEQ * DIM;            // attended_hypotheses

  float* ws   = (float*)d_ws;                            // 4 * B*SEQ floats = 512 KB
  float* mx_p = ws;
  float* sm_p = ws + (size_t)B * SEQ;
  float* mx_h = ws + (size_t)2 * B * SEQ;
  float* sm_h = ws + (size_t)3 * B * SEQ;

  dim3 grid(SEQ / ROWB, B);
  dim3 block(WAVES * 32);

  // prem -> hyp softmax stats, hyp -> prem softmax stats
  esim_stats_kernel<<<grid, block, 0, stream>>>(P, H, hm, mx_p, sm_p);
  esim_stats_kernel<<<grid, block, 0, stream>>>(H, P, pm, mx_h, sm_h);
  // attended_premises = P_attn @ H * pm ; attended_hypotheses = H_attn @ P * hm
  esim_attend_kernel<<<grid, block, 0, stream>>>(P, H, pm, hm, mx_p, sm_p, outP);
  esim_attend_kernel<<<grid, block, 0, stream>>>(H, P, hm, pm, mx_h, sm_h, outH);
}
